// LearnedSkipPredictor_78288663872348
// MI455X (gfx1250) — matrix operand order repaired and checked by reading, not compiled, not run on hardware
//
#include <hip/hip_runtime.h>
#include <math.h>
#include <stdint.h>

typedef __attribute__((ext_vector_type(16))) __bf16   v16bf;
typedef __attribute__((ext_vector_type(8)))  float    v8f;
typedef __attribute__((ext_vector_type(4)))  uint32_t u32x4;
typedef __attribute__((ext_vector_type(8)))  uint32_t u32x8;

#define B_   8
#define N_   4096
#define D_   1024
#define NC_  512
#define DH_  512     /* D/2: w1 output width      */
#define DQ_  256     /* D/4: ctx bottleneck width */
#define MIN_ACTIVE 819  /* int(4096*0.2) */

// ---------------------------------------------------------------------------
// TDM: async-load a 64x64 fp32 tile (row stride 1024 elements) into LDS.
// D# built per CDNA5 ISA (2-group form, 2D tile).
// ---------------------------------------------------------------------------
__device__ __forceinline__ void tdm_load_tile64x64(const float* gsrc,
                                                   uint32_t lds_byte_off)
{
    const uint64_t ga = (uint64_t)(uintptr_t)gsrc;
    u32x4 g0;
    g0[0] = 1u;                                   // count=1, user descriptor
    g0[1] = lds_byte_off;                         // lds_addr
    g0[2] = (uint32_t)ga;                         // global_addr[31:0]
    g0[3] = (uint32_t)((ga >> 32) & 0x01FFFFFFu)  // global_addr[56:32]
          | 0x80000000u;                          // type=2 ("image")
    u32x8 g1;
    g1[0] = (2u << 16);                           // data_size = 4 bytes
    g1[1] = ((uint32_t)D_ & 0xFFFFu) << 16;       // tensor_dim0 = 1024 (lo16)
    g1[2] = ((uint32_t)(B_ * N_) & 0xFFFFu) << 16;// dim0 hi=0 | tensor_dim1 lo16
    g1[3] = (64u << 16);                          // dim1 hi=0 | tile_dim0 = 64
    g1[4] = 64u;                                  // tile_dim1 = 64 | tile_dim2 = 0
    g1[5] = (uint32_t)D_;                         // tensor_dim0_stride = 1024
    g1[6] = 0u;                                   // stride hi | dim1_stride lo
    g1[7] = 0u;
    asm volatile("tensor_load_to_lds %0, %1" :: "s"(g0), "s"(g1) : "memory");
}

// ---------------------------------------------------------------------------
// Kernel 1: per-batch fused bias vector
// bias[b][j] = b1[j] + ctx_bn[b] @ w1[1024:1280, j] + t_emb[b] @ w1[1280:2304, j]
// ---------------------------------------------------------------------------
__global__ __launch_bounds__(256) void precompute_bias(
    const float* __restrict__ ctx_C, const int* __restrict__ t,
    const float* __restrict__ ctx_w, const float* __restrict__ ctx_b,
    const float* __restrict__ t_proj_w, const float* __restrict__ t_proj_b,
    const float* __restrict__ w1, const float* __restrict__ b1,
    float* __restrict__ bias_out)
{
    __shared__ float m[D_];
    __shared__ float embs[D_];
    __shared__ float cbn[DQ_];
    __shared__ float temb[D_];
    const int b = blockIdx.x, tid = threadIdx.x;

    for (int d = tid; d < D_; d += 256) {
        float s = 0.f;
        const float* p = ctx_C + (size_t)b * NC_ * D_ + d;
        for (int n = 0; n < NC_; ++n) s += p[(size_t)n * D_];
        m[d] = s * (1.0f / NC_);
    }
    const float tb = (float)t[b];
    for (int i = tid; i < 512; i += 256) {
        float fr = expf(-logf(10000.f) * (float)i / 511.f);
        float a  = tb * fr;
        embs[i]       = sinf(a);
        embs[i + 512] = cosf(a);
    }
    __syncthreads();

    if (tid < DQ_) {
        float s = ctx_b[tid];
        for (int d = 0; d < D_; ++d) s += m[d] * ctx_w[(size_t)d * DQ_ + tid];
        cbn[tid] = s;
    }
    for (int j = tid; j < D_; j += 256) {
        float s = t_proj_b[j];
        for (int d = 0; d < D_; ++d) s += embs[d] * t_proj_w[(size_t)d * D_ + j];
        temb[j] = s;
    }
    __syncthreads();

    for (int j = tid; j < DH_; j += 256) {
        float s = b1[j];
        for (int c = 0; c < DQ_; ++c) s += cbn[c]  * w1[(size_t)(D_ + c) * DH_ + j];
        for (int d = 0; d < D_;  ++d) s += temb[d] * w1[(size_t)(D_ + DQ_ + d) * DH_ + j];
        bias_out[b * DH_ + j] = s;
    }
}

// ---------------------------------------------------------------------------
// Kernel 2: pre-pack w1[0:1024, 0:512] into bf16 WMMA B-fragments.
//   n = nt*16 + (lane&15); k = kt*32 + (lane>>4)*16 + 2v + p  -> half index 2v+p
// ---------------------------------------------------------------------------
__global__ __launch_bounds__(256) void pack_w1_bf16(
    const float* __restrict__ w1, unsigned short* __restrict__ wp)
{
    const int g    = blockIdx.x * 256 + threadIdx.x; // 32768 total
    const int lane = g & 31;
    const int nt   = (g >> 5) & 31;
    const int kt   = g >> 10;
    const int n    = nt * 16 + (lane & 15);
    const int kb   = kt * 32 + (lane >> 4) * 16;
    __bf16* dst = (__bf16*)(wp + (size_t)(((kt * 32 + nt) * 32 + lane) * 16));
#pragma unroll
    for (int v = 0; v < 8; ++v) {
        dst[2 * v]     = (__bf16)w1[(size_t)(kb + 2 * v)     * DH_ + n];
        dst[2 * v + 1] = (__bf16)w1[(size_t)(kb + 2 * v + 1) * DH_ + n];
    }
}

// ---------------------------------------------------------------------------
// Kernel 3: WMMA GEMM. Block = 64 rows x 512 cols, 8 waves.
// Wave w: col-tiles w*4..w*4+3, all 4 row-tiles -> 16 accumulators.
// A staged to LDS by TDM (double-buffered 64x64 fp32 chunks), B from packed L2.
// ---------------------------------------------------------------------------
__global__ __launch_bounds__(256) void skip_gemm(
    const float* __restrict__ tokens, const float* __restrict__ bias_vec,
    const unsigned short* __restrict__ wp, const float* __restrict__ w2,
    const float* __restrict__ b2, float* __restrict__ scores_out)
{
    __shared__ float lds_a[2][64 * 64];   // 2 x 16 KiB fp32 A tiles
    __shared__ float rowsum[64];

    const int tid    = threadIdx.x;
    const int wave   = tid >> 5;
    const int lane   = tid & 31;
    const int lanelo = lane & 15;
    const int lh     = lane >> 4;
    const long r0    = (long)blockIdx.x * 64;     // flattened (B*N) row base
    const int batch  = (int)(r0 >> 12);           // / 4096 (blocks never straddle)

    v8f acc[4][4] = {};
    const float* gtile0 = tokens + (size_t)r0 * D_;

    if (wave == 0)
        tdm_load_tile64x64(gtile0, (uint32_t)(uintptr_t)&lds_a[0][0]);

    for (int kc = 0; kc < 16; ++kc) {
        if (wave == 0) {
            if (kc < 15) {
                tdm_load_tile64x64(gtile0 + (kc + 1) * 64,
                                   (uint32_t)(uintptr_t)&lds_a[(kc + 1) & 1][0]);
                __builtin_amdgcn_s_wait_tensorcnt((short)1);  // chunk kc done
            } else {
                __builtin_amdgcn_s_wait_tensorcnt((short)0);
            }
        }
        __syncthreads();                          // A chunk visible to all waves
        const float* abase = &lds_a[kc & 1][0];

#pragma unroll
        for (int ks = 0; ks < 2; ++ks) {
            const int ktg = kc * 2 + ks;          // global 32-wide K-step index
            v16bf afr[4];
#pragma unroll
            for (int mt = 0; mt < 4; ++mt) {      // A 16x32 bf16, contiguous k = 16*lh + s (matches B pack)
                const float* rp = abase + (mt * 16 + lanelo) * 64;
#pragma unroll
                for (int v = 0; v < 8; ++v) {
                    int k = ks * 32 + lh * 16 + v * 2;
                    float2 f2 = *(const float2*)(rp + k);
                    afr[mt][2 * v]     = (__bf16)f2.x;
                    afr[mt][2 * v + 1] = (__bf16)f2.y;
                }
            }
#pragma unroll
            for (int t4 = 0; t4 < 4; ++t4) {
                const int nt = wave * 4 + t4;
                v16bf bfrag = *(const v16bf*)(wp +
                    (size_t)(((ktg * 32 + nt) * 32 + lane) * 16));
#pragma unroll
                for (int mt = 0; mt < 4; ++mt)
                    acc[mt][t4] = __builtin_amdgcn_wmma_f32_16x16x32_bf16(
                        false, afr[mt], false, bfrag, (short)0, acc[mt][t4],
                        false, false);
            }
        }
        __syncthreads();                          // chunk consumed; TDM may reuse
    }

    // epilogue: bias + ReLU + h.w2 dot, reduced to 64 row scores
    if (tid < 64) rowsum[tid] = 0.f;
    __syncthreads();

#pragma unroll
    for (int mt = 0; mt < 4; ++mt) {
        float part[8];
#pragma unroll
        for (int r = 0; r < 8; ++r) part[r] = 0.f;
#pragma unroll
        for (int t4 = 0; t4 < 4; ++t4) {
            const int n  = wave * 64 + t4 * 16 + lanelo;
            const float bv  = bias_vec[batch * DH_ + n];
            const float w2v = w2[n];
#pragma unroll
            for (int r = 0; r < 8; ++r) {
                float h = acc[mt][t4][r] + bv;
                h = h > 0.f ? h : 0.f;
                part[r] += h * w2v;
            }
        }
#pragma unroll
        for (int m = 1; m <= 8; m <<= 1)
#pragma unroll
            for (int r = 0; r < 8; ++r)
                part[r] += __shfl_xor(part[r], m, 32);
        if (lanelo == 0) {
#pragma unroll
            for (int r = 0; r < 8; ++r)
                atomicAdd(&rowsum[mt * 16 + lh * 8 + r], part[r]);
        }
    }
    __syncthreads();
    if (tid < 64) {
        float x = rowsum[tid] + b2[0];
        scores_out[r0 + tid] = 1.f / (1.f + expf(-x));
    }
}

// ---------------------------------------------------------------------------
// Kernel 4: per-batch exact min-active floor (stable selection of the
// `deficit` lowest-scoring skipped tokens). One block per batch.
// ---------------------------------------------------------------------------
__global__ __launch_bounds__(256) void fix_skip(
    const float* __restrict__ scores, const int* __restrict__ rare,
    float* __restrict__ skip_out)
{
    __shared__ float sc[N_];
    __shared__ unsigned char fl[N_];
    __shared__ int cnt, c2;
    __shared__ int tcnt[256];
    const int b = blockIdx.x, tid = threadIdx.x;

    if (tid == 0) cnt = 0;
    __syncthreads();
    int myskip = 0;
    for (int i = tid; i < N_; i += 256) {
        float s = scores[b * N_ + i];
        int sk = (s > 0.5f) && (rare[b * N_ + i] == 0);
        sc[i] = s; fl[i] = (unsigned char)sk;
        myskip += sk;
    }
    atomicAdd(&cnt, myskip);
    __syncthreads();

    const int deficit = MIN_ACTIVE - (N_ - cnt);
    if (deficit > 0) {
        unsigned lo = 0u, hi = 0x3F800000u;       // scores in (0,1)
        while (lo < hi) {
            unsigned mid = lo + ((hi - lo) >> 1);
            if (tid == 0) c2 = 0;
            __syncthreads();
            int c = 0;
            for (int i = tid; i < N_; i += 256)
                if (fl[i] && __float_as_uint(sc[i]) <= mid) c++;
            atomicAdd(&c2, c);
            __syncthreads();
            if (c2 >= deficit) hi = mid; else lo = mid + 1;
            __syncthreads();
        }
        const unsigned thr = lo;                  // deficit-th smallest (bits)

        if (tid == 0) c2 = 0;
        __syncthreads();
        int c = 0;
        for (int i = tid; i < N_; i += 256)
            if (fl[i] && __float_as_uint(sc[i]) < thr) { fl[i] = 0; c++; }
        atomicAdd(&c2, c);
        __syncthreads();
        int rem = deficit - c2;                   // ties, stable by index

        const int base = tid * 16;
        int tc = 0;
        for (int i = base; i < base + 16; ++i)
            if (fl[i] && __float_as_uint(sc[i]) == thr) tc++;
        tcnt[tid] = tc;
        __syncthreads();
        if (tid == 0) {
            int run = 0;
            for (int i = 0; i < 256; ++i) { int v = tcnt[i]; tcnt[i] = run; run += v; }
        }
        __syncthreads();
        int p = tcnt[tid];
        for (int i = base; i < base + 16; ++i)
            if (fl[i] && __float_as_uint(sc[i]) == thr) { if (p < rem) fl[i] = 0; p++; }
        __syncthreads();
    }
    for (int i = tid; i < N_; i += 256)
        skip_out[b * N_ + i] = fl[i] ? 1.f : 0.f;
}

// ---------------------------------------------------------------------------
extern "C" void kernel_launch(void* const* d_in, const int* in_sizes, int n_in,
                              void* d_out, int out_size, void* d_ws, size_t ws_size,
                              hipStream_t stream) {
    const float* tokens   = (const float*)d_in[0];
    const float* ctx_C    = (const float*)d_in[1];
    const int*   t        = (const int*)d_in[2];
    const int*   rare     = (const int*)d_in[3];
    const float* ctx_w    = (const float*)d_in[4];
    const float* ctx_b    = (const float*)d_in[5];
    const float* t_proj_w = (const float*)d_in[6];
    const float* t_proj_b = (const float*)d_in[7];
    const float* w1       = (const float*)d_in[8];
    const float* b1       = (const float*)d_in[9];
    const float* w2       = (const float*)d_in[10];
    const float* b2       = (const float*)d_in[11];

    float* out_skip   = (float*)d_out;            // (B,N) skip as 0/1 floats
    float* out_scores = out_skip + B_ * N_;       // (B,N) scores

    char* ws = (char*)d_ws;
    float*          bias_ws = (float*)ws;                         // 16 KB
    unsigned short* wp      = (unsigned short*)(ws + 16 * 1024);  // 1 MiB bf16 pack

    precompute_bias<<<B_, 256, 0, stream>>>(ctx_C, t, ctx_w, ctx_b,
                                            t_proj_w, t_proj_b, w1, b1, bias_ws);
    pack_w1_bf16<<<128, 256, 0, stream>>>(w1, wp);
    skip_gemm<<<(B_ * N_) / 64, 256, 0, stream>>>(tokens, bias_ws, wp, w2, b2, out_scores);
    fix_skip<<<B_, 256, 0, stream>>>(out_scores, rare, out_skip);
}
